// PHGNS_19748259627133
// MI455X (gfx1250) — compile-verified, matrix operand order checked by reading, not run on hardware
//
#include <hip/hip_runtime.h>

typedef __bf16 bf16;
typedef bf16  v16bf __attribute__((ext_vector_type(16)));
typedef bf16  v8bf  __attribute__((ext_vector_type(8)));
typedef float v8f   __attribute__((ext_vector_type(8)));

#define EN 4096          // E == N == 4096
#define FF 8
#define MPITERS 5
#define DT_C 0.01f
#define LDS_PAD 40       // bf16 elems per 32-col LDS row (80B stride, conflict-free b128)
#define TILE_ELEMS (128 * LDS_PAD)

#define CDIV(a,b) (((a)+(b)-1)/(b))

// ---- CDNA5 async global->LDS copy (ASYNCcnt tracked), per-lane 16B ----
__device__ __forceinline__ void async_b128(unsigned lds_off, const bf16* g) {
    asm volatile("global_load_async_to_lds_b128 %0, %1, off"
                 :: "v"(lds_off), "v"(g) : "memory");
}
__device__ __forceinline__ void wait_async0() {
    asm volatile("s_wait_asynccnt 0" ::: "memory");
}

// ---------------------------------------------------------------------------
// WMMA GEMM: C[M x N](bf16) = epi( A[M x K](bf16) @ B[K x N](bf16) + bias )
// B is passed PRE-TRANSPOSED: BT is (N x K) row-major bf16 (fragment-ready).
// M,N % 128 == 0, K % 32 == 0.  256 thr = 8 wave32; wave: 16 rows x 128 cols.
// Double-buffered LDS tiles filled with global_load_async_to_lds_b128.
// ---------------------------------------------------------------------------
template<bool RELU, bool MASK>
__global__ __launch_bounds__(256)
void gemm_bf16(const bf16* __restrict__ A, const bf16* __restrict__ BT,
               bf16* __restrict__ C, const float* __restrict__ bias,
               const bf16* __restrict__ mask, int M, int N, int K)
{
    // single LDS allocation -> known offset 0; layout [A0|B0|A1|B1]
    __shared__ bf16 smem[4 * TILE_ELEMS];
    const int tid  = threadIdx.x;
    const int wave = tid >> 5;
    const int lane = tid & 31;
    const int lh   = lane >> 4;
    const int lm   = lane & 15;
    const int bm   = blockIdx.y * 128;
    const int bn   = blockIdx.x * 128;

    // 16B-granule mapping: 512 granules per 128x32 tile, 2 per thread
    const int r0 = tid >> 2,          o0 = (tid & 3) * 8;
    const int r1 = (tid + 256) >> 2,  o1 = ((tid + 256) & 3) * 8;
    const unsigned d0 = (unsigned)(r0 * LDS_PAD + o0) * 2u;
    const unsigned d1 = (unsigned)(r1 * LDS_PAD + o1) * 2u;
    const bf16* pA0 = A  + (size_t)(bm + r0) * K + o0;
    const bf16* pA1 = A  + (size_t)(bm + r1) * K + o1;
    const bf16* pB0 = BT + (size_t)(bn + r0) * K + o0;
    const bf16* pB1 = BT + (size_t)(bn + r1) * K + o1;
    const unsigned TB = (unsigned)TILE_ELEMS * 2u;   // tile bytes

    v8f acc[8];
#pragma unroll
    for (int i = 0; i < 8; ++i) {
        v8f z = {0.f,0.f,0.f,0.f,0.f,0.f,0.f,0.f};
        acc[i] = z;
    }

    auto issue = [&](int b) {
        unsigned ab = (unsigned)b * 2u * TB;
        unsigned bb = ab + TB;
        async_b128(ab + d0, pA0);
        async_b128(ab + d1, pA1);
        async_b128(bb + d0, pB0);
        async_b128(bb + d1, pB1);
        pA0 += 32; pA1 += 32; pB0 += 32; pB1 += 32;
    };

    issue(0);
    wait_async0();
    __syncthreads();

    int cur = 0;
    for (int kk = 0; kk < K; kk += 32) {
        if (kk + 32 < K) issue(cur ^ 1);

        const bf16* As = smem + (size_t)cur * 2 * TILE_ELEMS;
        const bf16* Bs = As + TILE_ELEMS;
        const int arow = wave * 16 + lm;
        v8bf a_lo = *(const v8bf*)(As + arow * LDS_PAD + lh * 8);
        v8bf a_hi = *(const v8bf*)(As + arow * LDS_PAD + 16 + lh * 8);
        v16bf afrag = __builtin_shufflevector(a_lo, a_hi,
                        0,1,2,3,4,5,6,7,8,9,10,11,12,13,14,15);
#pragma unroll
        for (int nt = 0; nt < 8; ++nt) {
            int bcol = nt * 16 + lm;
            v8bf b_lo = *(const v8bf*)(&Bs[bcol * LDS_PAD + lh * 8]);
            v8bf b_hi = *(const v8bf*)(&Bs[bcol * LDS_PAD + 16 + lh * 8]);
            v16bf bfrag = __builtin_shufflevector(b_lo, b_hi,
                            0,1,2,3,4,5,6,7,8,9,10,11,12,13,14,15);
            acc[nt] = __builtin_amdgcn_wmma_f32_16x16x32_bf16(
                          false, afrag, false, bfrag, (short)0, acc[nt],
                          false, false);
        }
        // LDS reads retired before the last WMMA (compiler waits dscnt);
        // now ensure next tile landed, then rendezvous before reuse.
        wait_async0();
        __syncthreads();
        cur ^= 1;
    }

#pragma unroll
    for (int nt = 0; nt < 8; ++nt) {
        int col = bn + nt * 16 + lm;
        float bv = bias ? bias[col] : 0.f;
#pragma unroll
        for (int v = 0; v < 8; ++v) {
            int row = bm + wave * 16 + lh * 8 + v;
            float val = acc[nt][v] + bv;
            if (RELU) val = val > 0.f ? val : 0.f;
            size_t idx = (size_t)row * N + col;
            if (MASK) { if (!((float)mask[idx] > 0.f)) val = 0.f; }
            C[idx] = (bf16)val;
        }
    }
}

// ---------------------------------------------------------------------------
// small helper kernels
// ---------------------------------------------------------------------------
__global__ void k_zero_f32(float* p, int n){ int i=blockIdx.x*256+threadIdx.x; if(i<n) p[i]=0.f; }
__global__ void k_zero_bf (bf16*  p, int n){ int i=blockIdx.x*256+threadIdx.x; if(i<n) p[i]=(bf16)0.f; }
__global__ void k_f2b(const float* s, bf16* d, int n){ int i=blockIdx.x*256+threadIdx.x; if(i<n) d[i]=(bf16)s[i]; }

__global__ void k_copy_pad(const float* __restrict__ s, bf16* __restrict__ d, int ncopy, int ntotal){
    int i = blockIdx.x*256+threadIdx.x;
    if (i < ntotal) d[i] = (i < ncopy) ? (bf16)s[i] : (bf16)0.f;
}
// pad A columns: dst[M x Kpad] from src[M x Kin]
__global__ void k_pad_cols(const float* __restrict__ s, bf16* __restrict__ d, int M, int Kin, int Kpad){
    int i = blockIdx.x*256+threadIdx.x;
    if (i >= M*Kpad) return;
    int m = i / Kpad, k = i - m*Kpad;
    d[i] = (k < Kin) ? (bf16)s[(size_t)m*Kin + k] : (bf16)0.f;
}
// WT[N x K] = W[K x N]^T  (f32 -> bf16)
__global__ void k_wt_t(const float* __restrict__ W, bf16* __restrict__ WT, int K, int N){
    int i = blockIdx.x*256+threadIdx.x;
    if (i >= K*N) return;
    int n = i / K, k = i - n*K;
    WT[i] = (bf16)W[(size_t)k*N + n];
}
// WT[N x Kpad] = pad(W[Kin x N])^T
__global__ void k_wt_t_pad(const float* __restrict__ W, bf16* __restrict__ WT, int Kin, int Kpad, int N){
    int i = blockIdx.x*256+threadIdx.x;
    if (i >= N*Kpad) return;
    int n = i / Kpad, k = i - n*Kpad;
    WT[i] = (k < Kin) ? (bf16)W[(size_t)k*N + n] : (bf16)0.f;
}
__global__ void k_concat_e(const bf16* __restrict__ e, const bf16* __restrict__ n,
                           const int* __restrict__ snd, const int* __restrict__ rcv,
                           bf16* __restrict__ ce){
    int idx = blockIdx.x*256+threadIdx.x;
    if (idx >= EN*768) return;
    int i = idx / 768, j = idx - i*768;
    bf16 v;
    if (j < 256)      v = e[(size_t)i*256 + j];
    else if (j < 512) v = n[(size_t)snd[i]*256 + (j-256)];
    else              v = n[(size_t)rcv[i]*256 + (j-512)];
    ce[idx] = v;
}
__global__ void k_concat_n(const bf16* __restrict__ n, const float* __restrict__ agg,
                           bf16* __restrict__ cn){
    int idx = blockIdx.x*256+threadIdx.x;
    if (idx >= EN*512) return;
    int i = idx >> 9, j = idx & 511;
    cn[idx] = (j < 256) ? n[(size_t)i*256 + j] : (bf16)agg[(size_t)i*256 + (j-256)];
}
__global__ void k_segsum(const bf16* __restrict__ e, const int* __restrict__ rcv,
                         float* __restrict__ agg){
    int idx = blockIdx.x*256+threadIdx.x;
    if (idx >= EN*256) return;
    int i = idx >> 8, f = idx & 255;
    atomicAdd(&agg[(size_t)rcv[i]*256 + f], (float)e[idx]);
}
__global__ void k_dec_grad(const bf16* __restrict__ hd, const float* __restrict__ W1,
                           bf16* __restrict__ dh){
    int idx = blockIdx.x*256+threadIdx.x;
    if (idx >= EN*256) return;
    int j = idx & 255;
    dh[idx] = ((float)hd[idx] > 0.f) ? (bf16)W1[j] : (bf16)0.f;
}
__global__ void k_grad_e_tot(const bf16* __restrict__ dep, const bf16* __restrict__ dcn,
                             const int* __restrict__ rcv, bf16* __restrict__ det){
    int idx = blockIdx.x*256+threadIdx.x;
    if (idx >= EN*256) return;
    int i = idx >> 8, j = idx & 255;
    det[idx] = (bf16)((float)dep[idx] + (float)dcn[(size_t)rcv[i]*512 + 256 + j]);
}
__global__ void k_init_dn(const bf16* __restrict__ dcn, float* __restrict__ dn){
    int idx = blockIdx.x*256+threadIdx.x;
    if (idx >= EN*256) return;
    int i = idx >> 8, j = idx & 255;
    dn[idx] = (float)dcn[(size_t)i*512 + j];
}
__global__ void k_scatter(const bf16* __restrict__ dce, const int* __restrict__ snd,
                          const int* __restrict__ rcv, float* __restrict__ dn,
                          bf16* __restrict__ dep){
    int idx = blockIdx.x*256+threadIdx.x;
    if (idx >= EN*256) return;
    int i = idx >> 8, j = idx & 255;
    atomicAdd(&dn[(size_t)snd[i]*256 + j], (float)dce[(size_t)i*768 + 256 + j]);
    atomicAdd(&dn[(size_t)rcv[i]*256 + j], (float)dce[(size_t)i*768 + 512 + j]);
    dep[idx] = dce[(size_t)i*768 + j];
}
__global__ void k_dx(const bf16* __restrict__ dh0, const float* __restrict__ W0,
                     float* __restrict__ dH){
    int i = blockIdx.x*256+threadIdx.x;
    if (i >= EN) return;
    float s = 0.f;
    for (int j = 0; j < 256; ++j) s += (float)dh0[(size_t)i*256 + j] * W0[j];
    dH[i] = s;
}
__global__ __launch_bounds__(256) void k_triJ(const float* __restrict__ Jk,
                                              const float* __restrict__ dH,
                                              float* __restrict__ f){
    int q = blockIdx.x, bi = 0, len = 32;
    while (q >= len) { q -= len; --len; ++bi; }
    int bj = bi + q;
    __shared__ float rs[128], cs[128];
    int t = threadIdx.x;
    if (t < 128) { rs[t] = 0.f; cs[t] = 0.f; }
    __syncthreads();
    int c = t & 127, r0 = t >> 7;
    int gj = bj*128 + c;
    float dhj = dH[gj], reg = 0.f;
    for (int r = r0; r < 128; r += 2) {
        int gi = bi*128 + r;
        float v = (gj >= gi) ? Jk[(size_t)gi*EN + gj] : 0.f;
        atomicAdd(&rs[r], v * dhj);
        reg += v * dH[gi];
    }
    atomicAdd(&cs[c], reg);
    __syncthreads();
    if (t < 128) {
        atomicAdd(&f[bi*128 + t],  rs[t]);
        atomicAdd(&f[bj*128 + t], -cs[t]);
    }
}
__global__ __launch_bounds__(256) void k_R1(const float* __restrict__ Rk,
                                            const float* __restrict__ dH,
                                            float* __restrict__ u){
    int q = blockIdx.x, a = 0, len = 32;
    while (q >= len) { q -= len; --len; ++a; }
    int bi = a + q, bj = a;          // bi >= bj
    __shared__ float cs[128];
    int t = threadIdx.x;
    if (t < 128) cs[t] = 0.f;
    __syncthreads();
    int c = t & 127, r0 = t >> 7;
    int gk = bj*128 + c;
    float reg = 0.f;
    for (int r = r0; r < 128; r += 2) {
        int gi = bi*128 + r;
        float v = (gi >= gk) ? Rk[(size_t)gi*EN + gk] : 0.f;
        reg += v * dH[gi];
    }
    atomicAdd(&cs[c], reg);
    __syncthreads();
    if (t < 128) atomicAdd(&u[bj*128 + t], cs[t]);
}
__global__ __launch_bounds__(256) void k_R2(const float* __restrict__ Rk,
                                            const float* __restrict__ u,
                                            float* __restrict__ f){
    int i = blockIdx.x;
    __shared__ float red[256];
    float s = 0.f;
    for (int k = threadIdx.x; k <= i; k += 256) s += Rk[(size_t)i*EN + k] * u[k];
    red[threadIdx.x] = s; __syncthreads();
    for (int o = 128; o > 0; o >>= 1) {
        if (threadIdx.x < o) red[threadIdx.x] += red[threadIdx.x + o];
        __syncthreads();
    }
    if (threadIdx.x == 0) f[i] -= red[0];
}
__global__ __launch_bounds__(256) void k_gmv(const float* __restrict__ g,
                                             const float* __restrict__ c,
                                             float* __restrict__ f){
    int i = blockIdx.x;
    __shared__ float red[256];
    float s = 0.f;
    for (int j = threadIdx.x; j < EN; j += 256) s += g[(size_t)i*EN + j] * c[j];
    red[threadIdx.x] = s; __syncthreads();
    for (int o = 128; o > 0; o >>= 1) {
        if (threadIdx.x < o) red[threadIdx.x] += red[threadIdx.x + o];
        __syncthreads();
    }
    if (threadIdx.x == 0) f[i] += red[0];
}
__global__ void k_final(const float* __restrict__ x0, const float* __restrict__ f,
                        float* __restrict__ out){
    int i = blockIdx.x*256+threadIdx.x;
    if (i < EN) out[i] = x0[i] + DT_C * f[i];
}

// ---------------------------------------------------------------------------
static void gemm(hipStream_t s, const bf16* A, const bf16* BT, bf16* C,
                 const float* bias, const bf16* mask, int M, int N, int K,
                 bool relu, bool maskf){
    dim3 g(N/128, M/128), b(256);
    if (relu)       gemm_bf16<true ,false><<<g,b,0,s>>>(A,BT,C,bias,nullptr,M,N,K);
    else if (maskf) gemm_bf16<false,true ><<<g,b,0,s>>>(A,BT,C,bias,mask,   M,N,K);
    else            gemm_bf16<false,false><<<g,b,0,s>>>(A,BT,C,bias,nullptr,M,N,K);
}

extern "C" void kernel_launch(void* const* d_in, const int* in_sizes, int n_in,
                              void* d_out, int out_size, void* d_ws, size_t ws_size,
                              hipStream_t stream) {
    (void)in_sizes; (void)n_in; (void)out_size; (void)ws_size;
    const float* edges0 = (const float*)d_in[0];
    const float* nodes  = (const float*)d_in[1];
    const float* ctrl   = (const float*)d_in[2];
    const float* eeW0 = (const float*)d_in[3];  const float* eeb0 = (const float*)d_in[4];
    const float* eeW1 = (const float*)d_in[5];  const float* eeb1 = (const float*)d_in[6];
    const float* enW0 = (const float*)d_in[7];  const float* enb0 = (const float*)d_in[8];
    const float* enW1 = (const float*)d_in[9];  const float* enb1 = (const float*)d_in[10];
    const float* meW0 = (const float*)d_in[11]; const float* meb0 = (const float*)d_in[12];
    const float* meW1 = (const float*)d_in[13]; const float* meb1 = (const float*)d_in[14];
    const float* mnW0 = (const float*)d_in[15]; const float* mnb0 = (const float*)d_in[16];
    const float* mnW1 = (const float*)d_in[17]; const float* mnb1 = (const float*)d_in[18];
    const float* deW0 = (const float*)d_in[19]; const float* deb0 = (const float*)d_in[20];
    const float* deW1 = (const float*)d_in[21];
    const float* Jk   = (const float*)d_in[23];
    const float* Rk   = (const float*)d_in[24];
    const float* gk   = (const float*)d_in[25];
    const int* snd    = (const int*)d_in[26];
    const int* rcv    = (const int*)d_in[27];
    float* out = (float*)d_out;

    char* wp = (char*)d_ws;
    auto alloc = [&](size_t bytes)->void* {
        void* r = (void*)wp; wp += (bytes + 255) & ~(size_t)255; return r;
    };
    // fragment-layout weights for FORWARD GEMMs: WT (N x K) bf16
    bf16* enW0pT = (bf16*)alloc(256*32*2);
    bf16* enW1T  = (bf16*)alloc(256*256*2);
    bf16* eeW0pT = (bf16*)alloc(256*32*2);
    bf16* eeW1T  = (bf16*)alloc(256*256*2);
    bf16* meW0T  = (bf16*)alloc(256*768*2);
    bf16* meW1T  = (bf16*)alloc(256*256*2);
    bf16* mnW0T  = (bf16*)alloc(256*512*2);
    bf16* mnW1T  = (bf16*)alloc(256*256*2);
    bf16* deW0T  = (bf16*)alloc(256*256*2);
    // plain bf16 weights = fragment layout of W^T, used by BACKWARD GEMMs
    bf16* eeW1b = (bf16*)alloc(256*256*2);
    bf16* meW0b = (bf16*)alloc(768*256*2);
    bf16* meW1b = (bf16*)alloc(256*256*2);
    bf16* mnW0b = (bf16*)alloc(512*256*2);
    bf16* mnW1b = (bf16*)alloc(256*256*2);
    bf16* deW0b = (bf16*)alloc(256*256*2);
    // activations
    bf16* nodes32 = (bf16*)alloc((size_t)EN*32*2);
    bf16* x32     = (bf16*)alloc((size_t)EN*32*2);
    bf16* scratch = (bf16*)alloc((size_t)EN*256*2);
    bf16* h0      = (bf16*)alloc((size_t)EN*256*2);
    bf16* n_cur   = (bf16*)alloc((size_t)EN*256*2);
    bf16* e_cur   = (bf16*)alloc((size_t)EN*256*2);
    bf16* he[MPITERS]; bf16* hn[MPITERS];
    for (int t = 0; t < MPITERS; ++t) { he[t]=(bf16*)alloc((size_t)EN*256*2); hn[t]=(bf16*)alloc((size_t)EN*256*2); }
    bf16* h_d  = (bf16*)alloc((size_t)EN*256*2);
    bf16* c_e  = (bf16*)alloc((size_t)EN*768*2);
    bf16* c_n  = (bf16*)alloc((size_t)EN*512*2);
    float* agg = (float*)alloc((size_t)EN*256*4);
    // backward
    bf16* d_h    = (bf16*)alloc((size_t)EN*256*2);
    bf16* d_ep   = (bf16*)alloc((size_t)EN*256*2);
    bf16* d_et   = (bf16*)alloc((size_t)EN*256*2);
    bf16* d_cn   = (bf16*)alloc((size_t)EN*512*2);
    bf16* d_ce   = (bf16*)alloc((size_t)EN*768*2);
    float* d_nf  = (float*)alloc((size_t)EN*256*4);
    bf16* d_nb   = (bf16*)alloc((size_t)EN*256*2);
    float* dH   = (float*)alloc(EN*4);
    float* uvec = (float*)alloc(EN*4);
    float* fbuf = (float*)alloc(EN*4);

    dim3 B(256);
    auto G1 = [](int n){ return dim3(CDIV(n,256)); };

    // ---- weight conversion: transposed (forward) + plain (backward) ----
    k_wt_t_pad<<<G1(256*32),B,0,stream>>>(enW0, enW0pT, FF, 32, 256);
    k_wt_t_pad<<<G1(256*32),B,0,stream>>>(eeW0, eeW0pT, 1,  32, 256);
    k_wt_t<<<G1(256*256),B,0,stream>>>(enW1, enW1T, 256, 256);
    k_wt_t<<<G1(256*256),B,0,stream>>>(eeW1, eeW1T, 256, 256);
    k_wt_t<<<G1(768*256),B,0,stream>>>(meW0, meW0T, 768, 256);
    k_wt_t<<<G1(256*256),B,0,stream>>>(meW1, meW1T, 256, 256);
    k_wt_t<<<G1(512*256),B,0,stream>>>(mnW0, mnW0T, 512, 256);
    k_wt_t<<<G1(256*256),B,0,stream>>>(mnW1, mnW1T, 256, 256);
    k_wt_t<<<G1(256*256),B,0,stream>>>(deW0, deW0T, 256, 256);
    k_copy_pad<<<G1(256*256),B,0,stream>>>(eeW1, eeW1b, 256*256, 256*256);
    k_copy_pad<<<G1(768*256),B,0,stream>>>(meW0, meW0b, 768*256, 768*256);
    k_copy_pad<<<G1(256*256),B,0,stream>>>(meW1, meW1b, 256*256, 256*256);
    k_copy_pad<<<G1(512*256),B,0,stream>>>(mnW0, mnW0b, 512*256, 512*256);
    k_copy_pad<<<G1(256*256),B,0,stream>>>(mnW1, mnW1b, 256*256, 256*256);
    k_copy_pad<<<G1(256*256),B,0,stream>>>(deW0, deW0b, 256*256, 256*256);
    // pad inputs (A-side K padded to 32)
    k_pad_cols<<<G1(EN*32),B,0,stream>>>(nodes,  nodes32, EN, FF, 32);
    k_pad_cols<<<G1(EN*32),B,0,stream>>>(edges0, x32,     EN, 1,  32);

    // ---- forward ----
    gemm(stream, nodes32, enW0pT, scratch, enb0, nullptr, EN, 256, 32, true, false);
    gemm(stream, scratch, enW1T,  n_cur,   enb1, nullptr, EN, 256, 256, false, false);
    gemm(stream, x32,     eeW0pT, h0,      eeb0, nullptr, EN, 256, 32, true, false);
    gemm(stream, h0,      eeW1T,  e_cur,   eeb1, nullptr, EN, 256, 256, false, false);
    for (int t = 0; t < MPITERS; ++t) {
        k_concat_e<<<G1(EN*768),B,0,stream>>>(e_cur, n_cur, snd, rcv, c_e);
        gemm(stream, c_e,   meW0T, he[t], meb0, nullptr, EN, 256, 768, true, false);
        gemm(stream, he[t], meW1T, e_cur, meb1, nullptr, EN, 256, 256, false, false);
        k_zero_f32<<<G1(EN*256),B,0,stream>>>(agg, EN*256);
        k_segsum<<<G1(EN*256),B,0,stream>>>(e_cur, rcv, agg);
        k_concat_n<<<G1(EN*512),B,0,stream>>>(n_cur, agg, c_n);
        gemm(stream, c_n,   mnW0T, hn[t], mnb0, nullptr, EN, 256, 512, true, false);
        gemm(stream, hn[t], mnW1T, n_cur, mnb1, nullptr, EN, 256, 256, false, false);
    }
    gemm(stream, e_cur, deW0T, h_d, deb0, nullptr, EN, 256, 256, true, false);

    // ---- backward (grad of sum(dec_e) wrt state) ----
    k_dec_grad<<<G1(EN*256),B,0,stream>>>(h_d, deW1, d_h);
    gemm(stream, d_h, deW0b, d_ep, nullptr, nullptr, EN, 256, 256, false, false);
    k_zero_bf<<<G1(EN*256),B,0,stream>>>(d_nb, EN*256);
    for (int t = MPITERS - 1; t >= 0; --t) {
        gemm(stream, d_nb, mnW1b, d_h,  nullptr, hn[t], EN, 256, 256, false, true);
        gemm(stream, d_h,  mnW0b, d_cn, nullptr, nullptr, EN, 512, 256, false, false);
        k_grad_e_tot<<<G1(EN*256),B,0,stream>>>(d_ep, d_cn, rcv, d_et);
        gemm(stream, d_et, meW1b, d_h,  nullptr, he[t], EN, 256, 256, false, true);
        gemm(stream, d_h,  meW0b, d_ce, nullptr, nullptr, EN, 768, 256, false, false);
        k_init_dn<<<G1(EN*256),B,0,stream>>>(d_cn, d_nf);
        k_scatter<<<G1(EN*256),B,0,stream>>>(d_ce, snd, rcv, d_nf, d_ep);
        k_f2b<<<G1(EN*256),B,0,stream>>>(d_nf, d_nb, EN*256);
    }
    gemm(stream, d_ep, eeW1b, d_h, nullptr, h0, EN, 256, 256, false, true);
    k_dx<<<G1(EN),B,0,stream>>>(d_h, eeW0, dH);

    // ---- (J - R) dH + g c ----
    k_zero_f32<<<G1(EN),B,0,stream>>>(fbuf, EN);
    k_zero_f32<<<G1(EN),B,0,stream>>>(uvec, EN);
    k_triJ<<<dim3(528),B,0,stream>>>(Jk, dH, fbuf);
    k_R1  <<<dim3(528),B,0,stream>>>(Rk, dH, uvec);
    k_R2  <<<dim3(EN), B,0,stream>>>(Rk, uvec, fbuf);
    k_gmv <<<dim3(EN), B,0,stream>>>(gk, ctrl, fbuf);
    k_final<<<G1(EN),B,0,stream>>>(edges0, fbuf, out);
}
